// SequentialClassifier_9414568313114
// MI455X (gfx1250) — matrix operand order repaired and checked by reading, not compiled, not run on hardware
//
#include <hip/hip_runtime.h>
#include <hip/hip_bf16.h>
#include <math.h>

// ---------------------------------------------------------------------------
// Autoregressive GRU decoder for MI455X (gfx1250, wave32, WMMA).
//   per step: G = [x_t|lab] @ W_ih^T and h @ W_hh^T via v_wmma_f32_16x16x32_bf16
//   then gates / argmax / label-embedding feedback in a per-row kernel.
// ---------------------------------------------------------------------------

typedef __attribute__((ext_vector_type(16))) __bf16 v16bf;
typedef __attribute__((ext_vector_type(4)))  __bf16 v4bf;
typedef __attribute__((ext_vector_type(8)))  float  v8f;
typedef __attribute__((ext_vector_type(4)))  float  v4f;

#define BATCH 128
#define SEQ   512
#define HID   1024
#define EMB   128
#define NLAB  32
#define GI_K  (HID + EMB)       // 1152
#define GH_K  HID               // 1024
#define GI_KC 36                // K-chunks (32 each) for gi
#define GH_KC 32                // K-chunks for gh
#define NT    (3 * HID / 16)    // 192 N-tiles per weight matrix
#define GW    (6 * HID)         // 6144: G = [gi(3072) | gh(3072)]
#define NPW   2                 // N-tiles per wave (block covers N=32)
#define NBLK  (NT / NPW)        // 96 blocks per matrix

// ---------------------------------------------------------------------------
// Pack fp32 weights into bf16 B-fragments (32x16 tile, WMMA lane layout):
//   lane 0..15  : N = lane,    element e -> K = kc*32 + (e<8 ? e : e+8)
//   lane 16..31 : N = lane-16, element e -> K = kc*32 + 8 + (e<8 ? e : e+8)
// Each lane's 16 bf16 values contiguous (one 32B load per lane).
// ---------------------------------------------------------------------------
__global__ __launch_bounds__(256) void pack_weights(
    const float* __restrict__ W_ih, const float* __restrict__ W_hh,
    __bf16* __restrict__ pack_gi, __bf16* __restrict__ pack_gh) {
  int idx = blockIdx.x * 256 + threadIdx.x;
  const int gi_total = NT * GI_KC * 512;
  const int gh_total = NT * GH_KC * 512;
  if (idx < gi_total) {
    int e    = idx & 15;
    int lane = (idx >> 4) & 31;
    int kc   = (idx >> 9) % GI_KC;
    int nt   = (idx >> 9) / GI_KC;
    int n    = nt * 16 + (lane & 15);
    int k    = kc * 32 + ((lane >= 16) ? 8 : 0) + ((e < 8) ? e : e + 8);
    pack_gi[idx] = (__bf16)W_ih[n * GI_K + k];
  } else if (idx - gi_total < gh_total) {
    int j    = idx - gi_total;
    int e    = j & 15;
    int lane = (j >> 4) & 31;
    int kc   = (j >> 9) % GH_KC;
    int nt   = (j >> 9) / GH_KC;
    int n    = nt * 16 + (lane & 15);
    int k    = kc * 32 + ((lane >= 16) ? 8 : 0) + ((e < 8) ? e : e + 8);
    pack_gh[j] = (__bf16)W_hh[n * GH_K + k];
  }
}

// h0 = 0, lab0 = emb[0] broadcast over batch
__global__ __launch_bounds__(256) void init_state(
    float* __restrict__ h, float* __restrict__ lab, const float* __restrict__ emb) {
  int idx = blockIdx.x * 256 + threadIdx.x;
  if (idx < BATCH * HID) h[idx] = 0.0f;
  if (idx < BATCH * EMB) lab[idx] = emb[idx & (EMB - 1)];  // emb[0][:]
}

// ---------------------------------------------------------------------------
// GEMM tile body: block = 256 threads = 8 wave32; M=128 (wave -> 16-row tile),
// N=32 (2 WMMA per wave), K streamed in 32-wide chunks.
// A element (m,k): k <  nkc0*32 -> src0[m*stride0 + k]
//                  k >= nkc0*32 -> src1[m*stride1 + (k - nkc0*32)]
// LDS double-buffered in WMMA A-fragment order:
//   lanes 0..15:  e<8 -> K=e,  e>=8 -> K=e+8;  lanes 16..31: +8 on K.
//   (any 4-aligned group of 4 K stays contiguous -> b128 load / b64 store)
// ---------------------------------------------------------------------------
__device__ __forceinline__ void gemm_body(
    const float* __restrict__ src0, size_t stride0, int nkc0,
    const float* __restrict__ src1, size_t stride1, int nkc,
    const __bf16* __restrict__ pb0, const __bf16* __restrict__ pb1,
    float* __restrict__ G, int ncolbase, __bf16* lds) {
  const int tid  = threadIdx.x;
  const int lane = tid & 31;
  const int w    = tid >> 5;

  // Per-thread staging geometry (invariant across K-chunks).
  int    ldsOff[4];
  size_t so0[4], so1[4];
#pragma unroll
  for (int i = 0; i < 4; ++i) {
    int idx = i * 256 + tid;         // 1024 groups of 4 floats
    int m   = idx >> 3;
    int k0  = (idx & 7) * 4;
    bool hi = ((k0 >= 8) && (k0 < 16)) || (k0 >= 24);
    int  e0 = (k0 < 8) ? k0 : ((k0 < 24) ? k0 - 8 : k0 - 16);
    int  ln = (m & 15) + (hi ? 16 : 0);
    int  mt = m >> 4;
    ldsOff[i] = (mt * 32 + ln) * 16 + e0;
    so0[i] = (size_t)m * stride0 + (size_t)k0;
    so1[i] = (size_t)m * stride1 + (size_t)k0;
  }

  auto stage = [&](int kc, int buf) {
    __bf16* dst = lds + (size_t)buf * (8 * 32 * 16);
    if (kc < nkc0) {
      const float* base = src0 + (size_t)kc * 32;
#pragma unroll
      for (int i = 0; i < 4; ++i) {
        v4f f = *reinterpret_cast<const v4f*>(base + so0[i]);
        v4bf b;
        b[0] = (__bf16)f[0]; b[1] = (__bf16)f[1];
        b[2] = (__bf16)f[2]; b[3] = (__bf16)f[3];
        *reinterpret_cast<v4bf*>(dst + ldsOff[i]) = b;
      }
    } else {
      const float* base = src1 + (size_t)(kc - nkc0) * 32;
#pragma unroll
      for (int i = 0; i < 4; ++i) {
        v4f f = *reinterpret_cast<const v4f*>(base + so1[i]);
        v4bf b;
        b[0] = (__bf16)f[0]; b[1] = (__bf16)f[1];
        b[2] = (__bf16)f[2]; b[3] = (__bf16)f[3];
        *reinterpret_cast<v4bf*>(dst + ldsOff[i]) = b;
      }
    }
  };

  v8f acc0 = {}, acc1 = {};
  stage(0, 0);
  __syncthreads();
  for (int kc = 0; kc < nkc; ++kc) {
    const int cur = kc & 1;
    v16bf a  = *reinterpret_cast<const v16bf*>(
        lds + (size_t)cur * (8 * 32 * 16) + (w * 32 + lane) * 16);
    v16bf b0 = *reinterpret_cast<const v16bf*>(pb0 + (size_t)kc * 512 + lane * 16);
    v16bf b1 = *reinterpret_cast<const v16bf*>(pb1 + (size_t)kc * 512 + lane * 16);
    if (kc + 1 < nkc) {
      __builtin_prefetch(pb0 + (size_t)(kc + 1) * 512 + lane * 16, 0, 3);
      __builtin_prefetch(pb1 + (size_t)(kc + 1) * 512 + lane * 16, 0, 3);
      stage(kc + 1, cur ^ 1);          // fill other buffer while WMMA runs
    }
    acc0 = __builtin_amdgcn_wmma_f32_16x16x32_bf16(
        false, a, false, b0, (short)0, acc0, false, false);
    acc1 = __builtin_amdgcn_wmma_f32_16x16x32_bf16(
        false, a, false, b1, (short)0, acc1, false, false);
    __syncthreads();                    // one barrier per chunk (double buffer)
  }

  // C layout: VGPR v, lanes 0..15 -> M=v; lanes 16..31 -> M=v+8; N = lane&15
  const int mofs = (lane >= 16) ? 8 : 0;
  const int nlo  = lane & 15;
#pragma unroll
  for (int v = 0; v < 8; ++v) {
    const size_t row = (size_t)(w * 16 + v + mofs) * GW;
    G[row + ncolbase + nlo]      = acc0[v];
    G[row + ncolbase + 16 + nlo] = acc1[v];
  }
}

// blocks 0..95: gi columns (A = [x_t | lab]); blocks 96..191: gh columns (A = h)
__global__ __launch_bounds__(256) void gemm_step(
    const float* __restrict__ hs, const float* __restrict__ lab,
    const float* __restrict__ h,
    const __bf16* __restrict__ pack_gi, const __bf16* __restrict__ pack_gh,
    float* __restrict__ G, int t) {
  __shared__ __align__(32) __bf16 ldsA[2][8][32][16];  // 16 KB double-buffered
  const int bb = blockIdx.x;
  if (bb < NBLK) {
    const int nt0 = bb * NPW;
    gemm_body(hs + (size_t)t * HID, (size_t)SEQ * HID, 32,
              lab, (size_t)EMB, GI_KC,
              pack_gi + (size_t)nt0 * GI_KC * 512,
              pack_gi + (size_t)(nt0 + 1) * GI_KC * 512,
              G, nt0 * 16, &ldsA[0][0][0][0]);
  } else {
    const int nt0 = (bb - NBLK) * NPW;
    gemm_body(h, (size_t)HID, GH_KC,
              h, (size_t)HID, GH_KC,
              pack_gh + (size_t)nt0 * GH_KC * 512,
              pack_gh + (size_t)(nt0 + 1) * GH_KC * 512,
              G, 3 * HID + nt0 * 16, &ldsA[0][0][0][0]);
  }
}

// ---------------------------------------------------------------------------
// update_step: one block per batch row. Gates -> h_new -> logits -> argmax ->
// feed emb[pred] back into lab. PyTorch GRU semantics (r,z,n gate order).
// ---------------------------------------------------------------------------
__global__ __launch_bounds__(256) void update_step(
    const float* __restrict__ G,
    const float* __restrict__ b_ih, const float* __restrict__ b_hh,
    const float* __restrict__ W_out, const float* __restrict__ b_out,
    const float* __restrict__ emb,
    float* __restrict__ h, float* __restrict__ lab,
    float* __restrict__ out, int t) {
  __shared__ float hn_lds[HID];
  __shared__ float part[NLAB][8];
  __shared__ int   pred_s;

  const int b   = blockIdx.x;
  const int tid = threadIdx.x;
  const float* Gb = G + (size_t)b * GW;

  for (int k = tid; k < HID; k += 256) {
    float ir  = Gb[k]           + b_ih[k];
    float iz  = Gb[HID + k]     + b_ih[HID + k];
    float in_ = Gb[2 * HID + k] + b_ih[2 * HID + k];
    float hr  = Gb[3 * HID + k] + b_hh[k];
    float hz  = Gb[4 * HID + k] + b_hh[HID + k];
    float hn  = Gb[5 * HID + k] + b_hh[2 * HID + k];
    float r = 1.0f / (1.0f + __expf(-(ir + hr)));
    float z = 1.0f / (1.0f + __expf(-(iz + hz)));
    float n = tanhf(in_ + r * hn);
    float hp = h[b * HID + k];
    float hnew = (1.0f - z) * n + z * hp;
    h[b * HID + k] = hnew;
    hn_lds[k] = hnew;
  }
  __syncthreads();

  {  // logits: 8 threads per label, 128-wide partial dot each
    int l = tid >> 3, sub = tid & 7;
    const float* wrow = W_out + l * HID + sub * 128;
    const float* hh   = hn_lds + sub * 128;
    float s = 0.0f;
    for (int k = 0; k < 128; ++k) s += hh[k] * wrow[k];
    part[l][sub] = s;
  }
  __syncthreads();

  if (tid < NLAB) {  // first wave32: finish logits + shuffle argmax
    float logit = b_out[tid];
    for (int j = 0; j < 8; ++j) logit += part[tid][j];
    out[((size_t)b * SEQ + t) * NLAB + tid] = logit;

    float v = logit;
    int   idx = tid;
    for (int off = 16; off > 0; off >>= 1) {
      float ov = __shfl_xor(v, off, 32);
      int   oi = __shfl_xor(idx, off, 32);
      if (ov > v || (ov == v && oi < idx)) { v = ov; idx = oi; }  // first-max tie
    }
    if (tid == 0) pred_s = idx;
  }
  __syncthreads();

  if (tid < EMB) lab[b * EMB + tid] = emb[pred_s * EMB + tid];
}

// ---------------------------------------------------------------------------
extern "C" void kernel_launch(void* const* d_in, const int* in_sizes, int n_in,
                              void* d_out, int out_size, void* d_ws, size_t ws_size,
                              hipStream_t stream) {
  const float* hs    = (const float*)d_in[0];
  const float* emb   = (const float*)d_in[1];
  const float* W_ih  = (const float*)d_in[2];
  const float* W_hh  = (const float*)d_in[3];
  const float* b_ih  = (const float*)d_in[4];
  const float* b_hh  = (const float*)d_in[5];
  const float* W_out = (const float*)d_in[6];
  const float* b_out = (const float*)d_in[7];
  float* out = (float*)d_out;

  // workspace layout (all re-written every call; no cross-call state)
  char* ws = (char*)d_ws;
  const size_t gi_elems = (size_t)NT * GI_KC * 512;   // 3,538,944
  const size_t gh_elems = (size_t)NT * GH_KC * 512;   // 3,145,728
  __bf16* pack_gi = (__bf16*)ws;
  __bf16* pack_gh = pack_gi + gi_elems;
  float*  h   = (float*)(ws + (gi_elems + gh_elems) * sizeof(__bf16));
  float*  lab = h + (size_t)BATCH * HID;
  float*  G   = lab + (size_t)BATCH * EMB;

  const int pack_total = (int)(gi_elems + gh_elems);
  pack_weights<<<(pack_total + 255) / 256, 256, 0, stream>>>(W_ih, W_hh, pack_gi, pack_gh);
  init_state<<<(BATCH * HID + 255) / 256, 256, 0, stream>>>(h, lab, emb);

  for (int t = 0; t < SEQ; ++t) {
    gemm_step<<<2 * NBLK, 256, 0, stream>>>(hs, lab, h, pack_gi, pack_gh, G, t);
    update_step<<<BATCH, 256, 0, stream>>>(G, b_ih, b_hh, W_out, b_out, emb, h, lab, out, t);
  }
}
